// EnhancedMambaLayer_3728031613106
// MI455X (gfx1250) — matrix operand order repaired and checked by reading, not compiled, not run on hardware
//
#include <hip/hip_runtime.h>
#include <hip/hip_bf16.h>
#include <math.h>

// ---------- types ----------
typedef __attribute__((ext_vector_type(16))) __bf16 bf16x16;
typedef __attribute__((ext_vector_type(8)))  __bf16 bf16x8;
typedef __attribute__((ext_vector_type(4)))  __bf16 bf16x4;
typedef __attribute__((ext_vector_type(2)))  __bf16 bf16x2;
typedef __attribute__((ext_vector_type(8)))  float  f32x8;
typedef __attribute__((ext_vector_type(4)))  float  f32x4;
typedef __attribute__((ext_vector_type(4)))  int    i32x4;

#define BB 2
#define LL 2048
#define DM 1024
#define DI 2048
#define DS 16
#define NH 8
#define DH 128
#define NE 4
#define DFF 4096
#define ROWS (BB*LL)   // 4096

#if __has_builtin(__builtin_amdgcn_global_load_async_to_lds_b128) && \
    __has_builtin(__builtin_amdgcn_s_wait_asynccnt)
#define ASYNC_CP 1
#else
#define ASYNC_CP 0
#endif

// round-to-nearest-even conversion (fill-time, once per element)
__device__ __forceinline__ __bf16 f2bf(float f) {
  unsigned int u = __builtin_bit_cast(unsigned int, f);
  unsigned int r = (u + 0x7FFFu + ((u >> 16) & 1u)) >> 16;
  return __builtin_bit_cast(__bf16, (unsigned short)r);
}
// truncating conversion (read-side hot path: 1 VALU op, v_mov_b16 .l,.h)
__device__ __forceinline__ __bf16 f2bf_t(float f) {
  unsigned int u = __builtin_bit_cast(unsigned int, f);
  return __builtin_bit_cast(__bf16, (unsigned short)(u >> 16));
}
__device__ __forceinline__ float siluf(float x) { return x / (1.f + __expf(-x)); }

// build a 16-element fragment from two contiguous 8-element LDS runs (2x ds_load_b128)
__device__ __forceinline__ bf16x16 frag2(const __bf16* lo, const __bf16* hi) {
  bf16x8 a = *(const bf16x8*)lo;
  bf16x8 b = *(const bf16x8*)hi;
  bf16x16 r;
#pragma unroll
  for (int i = 0; i < 8; i++) { r[i] = a[i]; r[i + 8] = b[i]; }
  return r;
}

// ---------- WMMA GEMM: C[M,N] = A[M,K] * B[K,N] row-major f32.
// Double-buffered: async A copies + B convert for tile t+1 overlap WMMAs of tile t.
// Requires M%128==0, K%32==0, N%4==0 (true for every call site). N guarded.
__global__ __launch_bounds__(256) void k_gemm(const float* __restrict__ A,
                                              const float* __restrict__ B,
                                              float* __restrict__ C,
                                              int M, int N, int K) {
  __shared__ float  sAraw[2][128][32];   // raw f32 A tiles (ASYNCcnt-tracked copies)
  __shared__ __bf16 sBt[2][128][32];     // B tiles transposed [n][k], bf16
  const int tid = threadIdx.x;
  const int wid = tid >> 5, lane = tid & 31;
  const int half = lane >> 4, ln = lane & 15;
  const int m0 = blockIdx.y * 128, n0 = blockIdx.x * 128;
  const int wrow = (wid >> 1) * 32, wcol = (wid & 1) * 64;

  f32x8 acc[2][4];
#pragma unroll
  for (int i = 0; i < 2; i++)
#pragma unroll
    for (int j = 0; j < 4; j++)
#pragma unroll
      for (int v = 0; v < 8; v++) acc[i][j][v] = 0.f;

  auto fillA = [&](int buf, int kk) {
#pragma unroll
    for (int it = 0; it < 4; it++) {
      int chunk = tid + it * 256;             // 1024 float4 chunks (128x32)
      int r = chunk >> 3, c4 = (chunk & 7) * 4;
      const float* gp = &A[(size_t)(m0 + r) * K + kk + c4];
#if ASYNC_CP
      __builtin_amdgcn_global_load_async_to_lds_b128(
          (i32x4*)gp, (i32x4*)&sAraw[buf][r][c4], 0, 0);
#else
      *(f32x4*)&sAraw[buf][r][c4] = *(const f32x4*)gp;
#endif
    }
  };
  auto fillB = [&](int buf, int kk) {
#pragma unroll
    for (int it = 0; it < 2; it++) {
      int chunk = tid + it * 256;             // 512 chunks (pairs of k-rows)
      int r2 = (chunk >> 5) * 2;
      int c4 = (chunk & 31) * 4;
      int gn = n0 + c4;
      int gnc = (gn < N) ? gn : (N - 4);      // clamp: N%4==0 so whole vec valid
      f32x4 va = *(const f32x4*)&B[(size_t)(kk + r2) * N + gnc];
      f32x4 vb = *(const f32x4*)&B[(size_t)(kk + r2 + 1) * N + gnc];
      if (gn >= N) {
#pragma unroll
        for (int q = 0; q < 4; q++) { va[q] = 0.f; vb[q] = 0.f; }
      }
#pragma unroll
      for (int q = 0; q < 4; q++) {
        bf16x2 p; p[0] = f2bf(va[q]); p[1] = f2bf(vb[q]);
        *(bf16x2*)&sBt[buf][c4 + q][r2] = p;
      }
    }
  };

  // prologue: stage tile 0
  fillA(0, 0);
  fillB(0, 0);
#if ASYNC_CP
  __builtin_amdgcn_s_wait_asynccnt(0);
#endif
  __syncthreads();

  for (int kk = 0; kk < K; kk += 32) {
    const int cur = (kk >> 5) & 1;
    const int nxt = cur ^ 1;
    const bool has_next = (kk + 32 < K);
    if (has_next) {           // overlap next-tile staging with this tile's WMMAs
      fillA(nxt, kk + 32);
      fillB(nxt, kk + 32);
      if (kk + 64 < K) {
        __builtin_prefetch(&A[(size_t)(m0 + (tid >> 1)) * K + kk + 64], 0, 1);
        __builtin_prefetch(&B[(size_t)(kk + 64 + (tid & 31)) * N + n0], 0, 1);
      }
    }

    // ---- fragments + 8 WMMAs per wave (buffer `cur`)
    bf16x16 bfrag[4];
#pragma unroll
    for (int j = 0; j < 4; j++) {
      const __bf16* bp = &sBt[cur][wcol + j * 16 + ln][8 * half];
      bfrag[j] = frag2(bp, bp + 16);
    }
#pragma unroll
    for (int i = 0; i < 2; i++) {
      const float* ar = &sAraw[cur][wrow + i * 16 + ln][0];
      f32x4 a0 = *(const f32x4*)&ar[8 * half];
      f32x4 a1 = *(const f32x4*)&ar[8 * half + 4];
      f32x4 a2 = *(const f32x4*)&ar[16 + 8 * half];
      f32x4 a3 = *(const f32x4*)&ar[16 + 8 * half + 4];
      bf16x16 af;
#pragma unroll
      for (int q = 0; q < 4; q++) {
        af[q]      = f2bf_t(a0[q]);
        af[4 + q]  = f2bf_t(a1[q]);
        af[8 + q]  = f2bf_t(a2[q]);
        af[12 + q] = f2bf_t(a3[q]);
      }
#pragma unroll
      for (int j = 0; j < 4; j++)
        acc[i][j] = __builtin_amdgcn_wmma_f32_16x16x32_bf16(
            false, af, false, bfrag[j], (short)0, acc[i][j], false, false);
    }
#if ASYNC_CP
    __builtin_amdgcn_s_wait_asynccnt(0);
#endif
    __syncthreads();
  }
#pragma unroll
  for (int i = 0; i < 2; i++)
#pragma unroll
    for (int j = 0; j < 4; j++)
#pragma unroll
      for (int v = 0; v < 8; v++) {
        int gm = m0 + wrow + i * 16 + v + 8 * half;
        int gn = n0 + wcol + j * 16 + ln;
        if (gn < N) C[(size_t)gm * N + gn] = acc[i][j][v];
      }
}

// ---------- flash attention: per-wave 16 queries, 32-key tiles ----------
__global__ __launch_bounds__(128) void k_attn(const float* __restrict__ qkv,
                                              float* __restrict__ ctx) {
  const int bh = blockIdx.y;
  const int b = bh / NH, h = bh % NH;
  const int wid = threadIdx.x >> 5, lane = threadIdx.x & 31;
  const int half = lane >> 4, ln = lane & 15;
  const int q0 = blockIdx.x * 64 + wid * 16;

  __shared__ __bf16 sK[32][DH];     // [key][dim] row-major
  __shared__ __bf16 sVt[DH][32];    // [dim][key] transposed
  __shared__ __bf16 sP[4][16][32];

  // Q fragments: 16x128 in 4 K-chunks of 32, vector loads (contiguous runs)
  bf16x16 qf[4];
  const float* qr = qkv + (size_t)(b * LL + q0 + ln) * 3072 + h * DH;
#pragma unroll
  for (int kc = 0; kc < 4; kc++) {
    f32x4 a0 = *(const f32x4*)&qr[kc * 32 + 8 * half];
    f32x4 a1 = *(const f32x4*)&qr[kc * 32 + 8 * half + 4];
    f32x4 a2 = *(const f32x4*)&qr[kc * 32 + 16 + 8 * half];
    f32x4 a3 = *(const f32x4*)&qr[kc * 32 + 16 + 8 * half + 4];
#pragma unroll
    for (int q = 0; q < 4; q++) {
      qf[kc][q]      = f2bf(a0[q]);
      qf[kc][4 + q]  = f2bf(a1[q]);
      qf[kc][8 + q]  = f2bf(a2[q]);
      qf[kc][12 + q] = f2bf(a3[q]);
    }
  }

  float m8[8], l8[8];
  f32x8 accO[8];
#pragma unroll
  for (int v = 0; v < 8; v++) { m8[v] = -1e30f; l8[v] = 0.f; }
#pragma unroll
  for (int nt = 0; nt < 8; nt++)
#pragma unroll
    for (int v = 0; v < 8; v++) accO[nt][v] = 0.f;

  for (int kt = 0; kt < LL / 32; kt++) {
    for (int idx = threadIdx.x; idx < 1024; idx += 128) {  // float4 chunks of 32x128
      int r = idx >> 5, c4 = (idx & 31) * 4;
      size_t row = (size_t)(b * LL + kt * 32 + r) * 3072;
      f32x4 kv = *(const f32x4*)&qkv[row + 1024 + h * DH + c4];
      f32x4 vv = *(const f32x4*)&qkv[row + 2048 + h * DH + c4];
      bf16x4 kp;
#pragma unroll
      for (int q = 0; q < 4; q++) kp[q] = f2bf(kv[q]);
      *(bf16x4*)&sK[r][c4] = kp;
#pragma unroll
      for (int q = 0; q < 4; q++) sVt[c4 + q][r] = f2bf(vv[q]);
    }
    __syncthreads();

    f32x8 s0, s1;
#pragma unroll
    for (int v = 0; v < 8; v++) { s0[v] = 0.f; s1[v] = 0.f; }
#pragma unroll
    for (int kc = 0; kc < 4; kc++) {
      const __bf16* kp0 = &sK[ln][kc * 32 + 8 * half];
      const __bf16* kp1 = &sK[16 + ln][kc * 32 + 8 * half];
      bf16x16 k0f = frag2(kp0, kp0 + 16);
      bf16x16 k1f = frag2(kp1, kp1 + 16);
      s0 = __builtin_amdgcn_wmma_f32_16x16x32_bf16(false, qf[kc], false, k0f, (short)0, s0, false, false);
      s1 = __builtin_amdgcn_wmma_f32_16x16x32_bf16(false, qf[kc], false, k1f, (short)0, s1, false, false);
    }

    // online softmax across 32 keys (16-lane half-group reductions)
#pragma unroll
    for (int v = 0; v < 8; v++) {
      float a = s0[v], c = s1[v];
      float mx = fmaxf(a, c);
#pragma unroll
      for (int m = 8; m >= 1; m >>= 1) mx = fmaxf(mx, __shfl_xor(mx, m, 32));
      float mn = fmaxf(m8[v], mx);
      float al = __expf(m8[v] - mn);
      float p0 = __expf(a - mn), p1 = __expf(c - mn);
      float ps = p0 + p1;
#pragma unroll
      for (int m = 8; m >= 1; m >>= 1) ps += __shfl_xor(ps, m, 32);
      l8[v] = l8[v] * al + ps;
      m8[v] = mn;
#pragma unroll
      for (int nt = 0; nt < 8; nt++) accO[nt][v] *= al;
      sP[wid][v + 8 * half][ln]      = f2bf(p0);
      sP[wid][v + 8 * half][16 + ln] = f2bf(p1);
    }

    const __bf16* pp = &sP[wid][ln][0];
    bf16x16 pf = frag2(pp + 8 * half, pp + 16 + 8 * half);
#pragma unroll
    for (int nt = 0; nt < 8; nt++) {
      const __bf16* vp = &sVt[nt * 16 + ln][8 * half];
      bf16x16 vf = frag2(vp, vp + 16);
      accO[nt] = __builtin_amdgcn_wmma_f32_16x16x32_bf16(false, pf, false, vf, (short)0, accO[nt], false, false);
    }
    __syncthreads();
  }

#pragma unroll
  for (int nt = 0; nt < 8; nt++)
#pragma unroll
    for (int v = 0; v < 8; v++) {
      int qrw = q0 + v + 8 * half;
      int col = h * DH + nt * 16 + ln;
      ctx[(size_t)(b * LL + qrw) * DM + col] = accO[nt][v] / l8[v];
    }
}

// ---------- elementwise kernels (vectorized b128 where shapes allow) ----------
__global__ void k_copy4(const float* __restrict__ in, float* __restrict__ out, int n4) {
  int i = blockIdx.x * blockDim.x + threadIdx.x;
  if (i < n4) *(f32x4*)&out[i * 4] = *(const f32x4*)&in[i * 4];
}
__global__ void k_zero4(float* __restrict__ p, int n4) {
  int i = blockIdx.x * blockDim.x + threadIdx.x;
  f32x4 z = {0.f, 0.f, 0.f, 0.f};
  if (i < n4) *(f32x4*)&p[i * 4] = z;
}
__global__ __launch_bounds__(256) void k_rmsnorm(const float* __restrict__ x,
                                                 const float* __restrict__ w,
                                                 float* __restrict__ out) {
  int row = blockIdx.x, tid = threadIdx.x;
  const float* xr = x + (size_t)row * DM;
  f32x4 v = *(const f32x4*)&xr[tid * 4];
  __shared__ float red[256];
  red[tid] = v.x * v.x + v.y * v.y + v.z * v.z + v.w * v.w;
  __syncthreads();
  for (int st = 128; st > 0; st >>= 1) { if (tid < st) red[tid] += red[tid + st]; __syncthreads(); }
  float r = rsqrtf(red[0] / (float)DM + 1e-6f);
  f32x4 w4 = *(const f32x4*)&w[tid * 4];
  f32x4 o = w4 * v * r;
  *(f32x4*)&out[(size_t)row * DM + tid * 4] = o;
}
__global__ void k_conv(const float* __restrict__ xin, const float* __restrict__ cw,
                       const float* __restrict__ cb, float* __restrict__ xc) {
  int i = blockIdx.x * blockDim.x + threadIdx.x;
  if (i >= ROWS * DI) return;
  int row = i >> 11, d = i & (DI - 1);
  int b = row >> 11, t = row & (LL - 1);
  float s = cb[d];
#pragma unroll
  for (int j = 0; j < 4; j++) {
    int tt = t + j - 2;
    if ((unsigned)tt < (unsigned)LL)
      s += cw[d * 4 + j] * xin[(size_t)(b * LL + tt) * (2 * DI) + d];
  }
  xc[i] = siluf(s);
}
__global__ void k_dt(float* __restrict__ sdt, float* __restrict__ sBm,
                     const float* __restrict__ bdt, const float* __restrict__ Aarr) {
  int i = blockIdx.x * blockDim.x + threadIdx.x;
  if (i >= ROWS * DS) return;
  int n = i & (DS - 1);
  float xv = sdt[i] + bdt[n];
  float dt = (xv > 20.f) ? xv : log1pf(__expf(xv));
  sdt[i] = __expf(dt * Aarr[n]);
  sBm[i] = dt * sBm[i];
}
__global__ void k_scan(const float* __restrict__ xc, const float* __restrict__ dec,
                       const float* __restrict__ dB, const float* __restrict__ cm,
                       const float* __restrict__ Dp, float* __restrict__ y) {
  int id = blockIdx.x * blockDim.x + threadIdx.x;
  if (id >= BB * DI) return;
  int b = id >> 11, d = id & (DI - 1);
  float h[DS];
#pragma unroll
  for (int n = 0; n < DS; n++) h[n] = 0.f;
  for (int t = 0; t < LL; t++) {
    int row = b * LL + t;
    float xv = xc[(size_t)row * DI + d];
    const f32x4* de = (const f32x4*)(dec + (size_t)row * DS);
    const f32x4* db = (const f32x4*)(dB + (size_t)row * DS);
    const f32x4* cc = (const f32x4*)(cm + (size_t)row * DS);
    float acc = 0.f;
#pragma unroll
    for (int q = 0; q < 4; q++) {
      f32x4 d4 = de[q], b4 = db[q], c4 = cc[q];
#pragma unroll
      for (int n = 0; n < 4; n++) {
        h[q * 4 + n] = d4[n] * h[q * 4 + n] + b4[n] * xv;
        acc += h[q * 4 + n] * c4[n];
      }
    }
    y[(size_t)row * DI + d] = acc + Dp[d] * xv;
  }
}
__global__ void k_yz4(float* __restrict__ y, const float* __restrict__ xin, int n4) {
  int i = blockIdx.x * blockDim.x + threadIdx.x;
  if (i >= n4) return;
  int i4 = i * 4;
  int row = i4 >> 11, d = i4 & (DI - 1);
  f32x4 z = *(const f32x4*)&xin[(size_t)row * (2 * DI) + DI + d];
  f32x4 yy = *(f32x4*)&y[i4];
#pragma unroll
  for (int q = 0; q < 4; q++) yy[q] *= siluf(z[q]);
  *(f32x4*)&y[i4] = yy;
}
__global__ void k_resadd4(float* __restrict__ res, const float* __restrict__ src,
                          const float* __restrict__ ls, int n4) {
  int i = blockIdx.x * blockDim.x + threadIdx.x;
  if (i >= n4) return;
  int i4 = i * 4;
  f32x4 l = *(const f32x4*)&ls[i4 & (DM - 1)];
  f32x4 r = *(f32x4*)&res[i4];
  f32x4 s = *(const f32x4*)&src[i4];
  *(f32x4*)&res[i4] = r + l * s;
}
__global__ void k_rope(const float* __restrict__ in, float* __restrict__ out,
                       const float* __restrict__ ascale) {
  int i = blockIdx.x * blockDim.x + threadIdx.x;
  if (i >= ROWS * 3072) return;
  int row = i / 3072, c = i - row * 3072;
  if (c >= 2048) { out[i] = in[i]; return; }
  int sec = (c >= 1024) ? 1 : 0;
  int cc = c - sec * 1024;
  int h = cc >> 7, d = cc & 127;
  int fi = d & 63;
  float inv = __powf(10000.f, -(float)fi / 64.f);
  float ang = (float)(row & (LL - 1)) * inv;
  const float* base = in + (size_t)row * 3072 + sec * 1024 + h * DH;
  float xd = base[d];
  float rot = (d < 64) ? -base[2 * d + 1] : base[2 * (d - 64)];
  float val = xd * __cosf(ang) + rot * __sinf(ang);
  if (sec == 0) val *= sqrtf((float)DH) * ascale[h];
  out[i] = val;
}
__global__ void k_relu_bias(float* __restrict__ g, const float* __restrict__ bg, int n) {
  int i = blockIdx.x * blockDim.x + threadIdx.x;
  if (i < n) g[i] = fmaxf(g[i] + bg[i & 511], 0.f);
}
__global__ void k_gate(const float* __restrict__ logits, float* __restrict__ topw,
                       int* __restrict__ topi, float* __restrict__ gmean) {
  int row = blockIdx.x * blockDim.x + threadIdx.x;
  if (row >= ROWS) return;
  float lg[NE], gs[NE];
  float mx = -1e30f;
#pragma unroll
  for (int e = 0; e < NE; e++) { lg[e] = logits[row * NE + e]; mx = fmaxf(mx, lg[e]); }
  float den = 0.f;
#pragma unroll
  for (int e = 0; e < NE; e++) { gs[e] = __expf(lg[e] - mx); den += gs[e]; }
#pragma unroll
  for (int e = 0; e < NE; e++) {
    gs[e] /= den;
    atomicAdd(&gmean[e], gs[e] * (1.f / (float)ROWS));
  }
  int i0 = 0;
#pragma unroll
  for (int e = 1; e < NE; e++) if (gs[e] > gs[i0]) i0 = e;
  int i1 = (i0 == 0) ? 1 : 0;
#pragma unroll
  for (int e = 0; e < NE; e++) if (e != i0 && gs[e] > gs[i1]) i1 = e;
  float t1 = __expf(gs[i1] - gs[i0]);
  float dn = 1.f + t1;
  topw[row * 2] = 1.f / dn; topw[row * 2 + 1] = t1 / dn;
  topi[row * 2] = i0; topi[row * 2 + 1] = i1;
}
__global__ void k_bal(const float* __restrict__ gmean, float* __restrict__ out) {
  if (threadIdx.x == 0) {
    float s = 0.f;
#pragma unroll
    for (int e = 0; e < NE; e++) { float g = gmean[e]; s += g * __logf(g + 1e-8f); }
    out[0] = (float)NE * s;
  }
}
__global__ void k_silumul4(float* __restrict__ a, const float* __restrict__ b, int n4) {
  int i = blockIdx.x * blockDim.x + threadIdx.x;
  if (i >= n4) return;
  int i4 = i * 4;
  f32x4 av = *(f32x4*)&a[i4];
  f32x4 bv = *(const f32x4*)&b[i4];
#pragma unroll
  for (int q = 0; q < 4; q++) av[q] = siluf(av[q]) * bv[q];
  *(f32x4*)&a[i4] = av;
}
__global__ void k_combine4(const float* __restrict__ eo, const float* __restrict__ topw,
                           const int* __restrict__ topi, int e, float* __restrict__ acc) {
  int i = blockIdx.x * blockDim.x + threadIdx.x;
  if (i >= ROWS * DM / 4) return;
  int i4 = i * 4;
  int row = i4 >> 10;
  float w = 0.f;
  if (topi[row * 2] == e) w += topw[row * 2];
  if (topi[row * 2 + 1] == e) w += topw[row * 2 + 1];
  f32x4 a = *(f32x4*)&acc[i4];
  f32x4 s = *(const f32x4*)&eo[i4];
  *(f32x4*)&acc[i4] = a + w * s;
}

// ---------- launch ----------
static inline dim3 g1d(int n, int b) { return dim3((n + b - 1) / b); }

extern "C" void kernel_launch(void* const* d_in, const int* in_sizes, int n_in,
                              void* d_out, int out_size, void* d_ws, size_t ws_size,
                              hipStream_t stream) {
  (void)in_sizes; (void)n_in; (void)out_size; (void)ws_size;
  const float* x      = (const float*)d_in[0];
  const float* rms1   = (const float*)d_in[1];
  const float* rms2   = (const float*)d_in[2];
  const float* rms3   = (const float*)d_in[3];
  const float* ls1    = (const float*)d_in[4];
  const float* ls2    = (const float*)d_in[5];
  const float* ls3    = (const float*)d_in[6];
  const float* W_in   = (const float*)d_in[7];
  const float* conv_w = (const float*)d_in[8];
  const float* conv_b = (const float*)d_in[9];
  const float* W_B    = (const float*)d_in[10];
  const float* W_C    = (const float*)d_in[11];
  const float* W_dt   = (const float*)d_in[12];
  const float* b_dt   = (const float*)d_in[13];
  const float* W_out  = (const float*)d_in[14];
  const float* Aarr   = (const float*)d_in[15];
  const float* Dp     = (const float*)d_in[16];
  const float* W_qkv  = (const float*)d_in[17];
  const float* W_o    = (const float*)d_in[18];
  const float* ascale = (const float*)d_in[19];
  const float* Wg1    = (const float*)d_in[20];
  const float* bg1    = (const float*)d_in[21];
  const float* Wg2    = (const float*)d_in[22];
  const float* w1     = (const float*)d_in[23];
  const float* w2     = (const float*)d_in[24];
  const float* w3     = (const float*)d_in[25];

  float* res = (float*)d_out;
  float* bal = res + (size_t)ROWS * DM;

  float* W = (float*)d_ws;
  size_t off = 0;
  float* f_norm = W + off; off += (size_t)ROWS * DM;
  float* big0   = W + off; off += (size_t)ROWS * DFF;
  float* big1   = W + off; off += (size_t)ROWS * DFF;
  float* mid0   = W + off; off += (size_t)ROWS * DI;
  float* mid1   = W + off; off += (size_t)ROWS * DI;
  float* outA   = W + off; off += (size_t)ROWS * DM;
  float* outB   = W + off; off += (size_t)ROWS * DM;
  float* sdt    = W + off; off += (size_t)ROWS * DS;
  float* sBm    = W + off; off += (size_t)ROWS * DS;
  float* sCm    = W + off; off += (size_t)ROWS * DS;
  float* g1buf  = W + off; off += (size_t)ROWS * 512;
  float* logits = W + off; off += (size_t)ROWS * NE;
  float* topw   = W + off; off += (size_t)ROWS * 2;
  int*   topi   = (int*)(W + off); off += (size_t)ROWS * 2;
  float* gmean  = W + off; off += 4;

  const int EW = 256;
  dim3 blk(EW);
  dim3 gblk256(256);
  dim3 gblk128(128);
  auto gemm_grid = [](int M, int N) { return dim3((N + 127) / 128, M / 128); };

  k_copy4<<<g1d(ROWS * DM / 4, EW), blk, 0, stream>>>(x, res, ROWS * DM / 4);

  // ---- Mamba ----
  k_rmsnorm<<<dim3(ROWS), dim3(256), 0, stream>>>(res, rms1, f_norm);
  k_gemm<<<gemm_grid(ROWS, 2 * DI), gblk256, 0, stream>>>(f_norm, W_in, big0, ROWS, 2 * DI, DM);
  k_conv<<<g1d(ROWS * DI, EW), blk, 0, stream>>>(big0, conv_w, conv_b, mid0);
  k_gemm<<<gemm_grid(ROWS, DS), gblk256, 0, stream>>>(mid0, W_dt, sdt, ROWS, DS, DI);
  k_gemm<<<gemm_grid(ROWS, DS), gblk256, 0, stream>>>(mid0, W_B, sBm, ROWS, DS, DI);
  k_gemm<<<gemm_grid(ROWS, DS), gblk256, 0, stream>>>(mid0, W_C, sCm, ROWS, DS, DI);
  k_dt<<<g1d(ROWS * DS, EW), blk, 0, stream>>>(sdt, sBm, b_dt, Aarr);
  k_scan<<<g1d(BB * DI, EW), blk, 0, stream>>>(mid0, sdt, sBm, sCm, Dp, mid1);
  k_yz4<<<g1d(ROWS * DI / 4, EW), blk, 0, stream>>>(mid1, big0, ROWS * DI / 4);
  k_gemm<<<gemm_grid(ROWS, DM), gblk256, 0, stream>>>(mid1, W_out, outA, ROWS, DM, DI);
  k_resadd4<<<g1d(ROWS * DM / 4, EW), blk, 0, stream>>>(res, outA, ls1, ROWS * DM / 4);

  // ---- Attention ----
  k_rmsnorm<<<dim3(ROWS), dim3(256), 0, stream>>>(res, rms2, f_norm);
  k_gemm<<<gemm_grid(ROWS, 3 * DM), gblk256, 0, stream>>>(f_norm, W_qkv, big0, ROWS, 3 * DM, DM);
  k_rope<<<g1d(ROWS * 3072, EW), blk, 0, stream>>>(big0, big1, ascale);
  k_attn<<<dim3(LL / 64, BB * NH), gblk128, 0, stream>>>(big1, outB);
  k_gemm<<<gemm_grid(ROWS, DM), gblk256, 0, stream>>>(outB, W_o, outA, ROWS, DM, DM);
  k_resadd4<<<g1d(ROWS * DM / 4, EW), blk, 0, stream>>>(res, outA, ls2, ROWS * DM / 4);

  // ---- MoE ----
  k_rmsnorm<<<dim3(ROWS), dim3(256), 0, stream>>>(res, rms3, f_norm);
  k_gemm<<<gemm_grid(ROWS, 512), gblk256, 0, stream>>>(f_norm, Wg1, g1buf, ROWS, 512, DM);
  k_relu_bias<<<g1d(ROWS * 512, EW), blk, 0, stream>>>(g1buf, bg1, ROWS * 512);
  k_gemm<<<gemm_grid(ROWS, NE), gblk256, 0, stream>>>(g1buf, Wg2, logits, ROWS, NE, 512);
  k_zero4<<<dim3(1), dim3(32), 0, stream>>>(gmean, 1);
  k_gate<<<g1d(ROWS, EW), blk, 0, stream>>>(logits, topw, topi, gmean);
  k_bal<<<dim3(1), dim3(32), 0, stream>>>(gmean, bal);
  k_zero4<<<g1d(ROWS * DM / 4, EW), blk, 0, stream>>>(outB, ROWS * DM / 4);
  for (int e = 0; e < NE; e++) {
    const float* w1e = w1 + (size_t)e * DM * DFF;
    const float* w2e = w2 + (size_t)e * DM * DFF;
    const float* w3e = w3 + (size_t)e * DFF * DM;
    k_gemm<<<gemm_grid(ROWS, DFF), gblk256, 0, stream>>>(f_norm, w1e, big0, ROWS, DFF, DM);
    k_gemm<<<gemm_grid(ROWS, DFF), gblk256, 0, stream>>>(f_norm, w2e, big1, ROWS, DFF, DM);
    k_silumul4<<<g1d(ROWS * DFF / 4, EW), blk, 0, stream>>>(big0, big1, ROWS * DFF / 4);
    k_gemm<<<gemm_grid(ROWS, DM), gblk256, 0, stream>>>(big0, w3e, outA, ROWS, DM, DFF);
    k_combine4<<<g1d(ROWS * DM / 4, EW), blk, 0, stream>>>(outA, topw, topi, e, outB);
  }
  k_resadd4<<<g1d(ROWS * DM / 4, EW), blk, 0, stream>>>(res, outB, ls3, ROWS * DM / 4);
}
